// MultiHeadAttention_76106820485310
// MI455X (gfx1250) — matrix operand order repaired and checked by reading, not compiled, not run on hardware
//
#include <hip/hip_runtime.h>
#include <hip/hip_bf16.h>

// ---------------------------------------------------------------------------
// MI455X / gfx1250 multi-head attention block (post-LN), WMMA f16->f32 path.
// Dense GEMMs: 16x64 wave tiles, B staged per-block through LDS.
// Attention: flash-style online softmax; K/V chunks staged per-block in LDS
// (4 waves share), WMMA fed from ds_load_b128.
// ---------------------------------------------------------------------------

typedef __attribute__((ext_vector_type(16))) _Float16 v16h;
typedef __attribute__((ext_vector_type(8)))  _Float16 v8h;
typedef __attribute__((ext_vector_type(8)))  float    v8f;

#define D_MODEL 1024
#define N_HEAD  16
#define DK      64
#define NSEQ    2048
#define BATCH   2
#define MTOT    (BATCH * NSEQ)          // 4096 rows total
#define BSTRIDE 40                      // LDS row stride (halfs), 32-wide rows
#define KSTRIDE 72                      // LDS row stride (halfs), 64-wide rows

static __device__ __forceinline__ v8f wmma16x16x32(v16h a, v16h b, v8f c) {
  // v_wmma_f32_16x16x32_f16  D = A*B + C  (f32 accumulate)
  return __builtin_amdgcn_wmma_f32_16x16x32_f16(false, a, false, b, (short)0, c,
                                                false, false);
}

union AFrag { v16h v; v8h h[2]; };

// ---------------------------------------------------------------------------
// 1) elementwise f32 -> f16 convert
// ---------------------------------------------------------------------------
__global__ void k_cvt_f16(const float* __restrict__ in, _Float16* __restrict__ out,
                          int n) {
  int i = blockIdx.x * blockDim.x + threadIdx.x;
  if (i < n) out[i] = (_Float16)in[i];
}

// ---------------------------------------------------------------------------
// 2) W[K][N] f32  ->  Wt[N][K] f16   (32x32 LDS tiles, K=N=1024)
// ---------------------------------------------------------------------------
__global__ void k_transpose_cvt(const float* __restrict__ w,
                                _Float16* __restrict__ wt, int K, int N) {
  __shared__ float tile[32][33];
  int k0 = blockIdx.x * 32, n0 = blockIdx.y * 32;
  int tx = threadIdx.x & 31, ty = threadIdx.x >> 5;   // 32 x 8
  for (int r = ty; r < 32; r += 8)
    tile[r][tx] = w[(size_t)(k0 + r) * N + n0 + tx];
  __syncthreads();
  for (int r = ty; r < 32; r += 8)
    wt[(size_t)(n0 + r) * K + k0 + tx] = (_Float16)tile[tx][r];
}

// ---------------------------------------------------------------------------
// 3) Projection GEMM:  C[4096,1024] = A[4096,1024] * W  (W given as Wt[N][K])
//    Block = 8 waves, tile 128x64; B chunk staged in LDS per 32-wide K step.
//      mode 0: head-major  [b, h, seq, dk]   (Q with scale=1/8, K with scale=1)
//      mode 1: transposed  [b, h, dk, seq]   (V)
// ---------------------------------------------------------------------------
__global__ __launch_bounds__(256)
void k_proj_gemm(const _Float16* __restrict__ A,
                 const _Float16* __restrict__ Bt,
                 _Float16* __restrict__ out,
                 float scale, int mode) {
  __shared__ __align__(16) _Float16 bsh[64 * BSTRIDE];
  int tid  = threadIdx.x;
  int wave = tid >> 5;
  int lane = tid & 31;
  int hi   = lane >> 4;          // lane half: 0 -> lanes 0-15, 1 -> lanes 16-31
  int l16  = lane & 15;
  int m0   = blockIdx.x * 128 + wave * 16;
  int n0   = blockIdx.y * 64;

  const _Float16* arow = A + (size_t)(m0 + l16) * D_MODEL;

  // B staging: thread tid moves Bt[n0 + tid/4][kb + (tid%4)*8 .. +7]
  int srow = tid >> 2;
  int scol = (tid & 3) * 8;
  const _Float16* bsrc = Bt + (size_t)(n0 + srow) * D_MODEL + scol;
  _Float16*       bdst = bsh + srow * BSTRIDE + scol;

  v8f acc[4] = {};
  for (int kb = 0; kb < D_MODEL; kb += 32) {
    v8h stg = *(const v8h*)(bsrc + kb);     // overlaps previous chunk compute
    __syncthreads();                        // previous chunk fully consumed
    *(v8h*)bdst = stg;
    __syncthreads();                        // chunk visible to all waves

    AFrag a;                                // shared across the 4 N-tiles
    a.h[0] = *(const v8h*)(arow + kb + hi * 16);       // K = kb + 16*hi .. +7
    a.h[1] = *(const v8h*)(arow + kb + hi * 16 + 8);   // K = kb+16*hi+8 .. +15
#pragma unroll
    for (int t = 0; t < 4; ++t) {
      const _Float16* bls = bsh + (t * 16 + l16) * BSTRIDE + hi * 16;
      AFrag bf;                             // B fragment via 2x ds_load_b128
      bf.h[0] = *(const v8h*)(bls);
      bf.h[1] = *(const v8h*)(bls + 8);
      acc[t] = wmma16x16x32(a.v, bf.v, acc[t]);
    }
  }

#pragma unroll
  for (int t = 0; t < 4; ++t) {
#pragma unroll
    for (int r = 0; r < 8; ++r) {
      int row  = m0 + r + hi * 8;           // C layout: VGPR r <-> M = r + 8*hi
      int col  = n0 + t * 16 + l16;
      int bidx = row >> 11, seq = row & (NSEQ - 1);
      int h    = col >> 6,  dk  = col & (DK - 1);
      float vv = acc[t][r] * scale;
      size_t idx;
      if (mode == 0)
        idx = (((size_t)bidx * N_HEAD + h) * NSEQ + seq) * DK + dk;
      else
        idx = (((size_t)bidx * N_HEAD + h) * DK + dk) * NSEQ + seq;
      out[idx] = (_Float16)vv;
    }
  }
}

// ---------------------------------------------------------------------------
// 4) Flash attention.  grid = (NSEQ/16/4, BATCH*N_HEAD), block = 128 (4 waves).
//    Each wave owns one 16-row q tile.  Per 32-wide kv chunk the block stages
//    K (32x64) and V (64x32) in LDS, shared by all 4 waves; online softmax
//    per VGPR slot; P staged through per-wave LDS to form the A-fragment.
// ---------------------------------------------------------------------------
__global__ __launch_bounds__(128)
void k_attention(const _Float16* __restrict__ Qh,   // [bh][n][64] (prescaled)
                 const _Float16* __restrict__ Kh,   // [bh][n][64]
                 const _Float16* __restrict__ Vt,   // [bh][64][n]
                 _Float16* __restrict__ ctx) {      // [b][n][h*64+dv]
  __shared__ __align__(16) _Float16 ksh[32 * KSTRIDE];   // K chunk, row=kv row
  __shared__ __align__(16) _Float16 vsh[64 * BSTRIDE];   // V chunk, row=dv
  __shared__ __align__(16) _Float16 psh[4 * 16 * 32];    // per-wave P staging
  int tid  = threadIdx.x;
  int wave = tid >> 5;
  int lane = tid & 31;
  int hi   = lane >> 4;
  int l16  = lane & 15;
  int bh   = blockIdx.y;                         // b*16 + h
  int qt   = blockIdx.x * 4 + wave;              // q tile index (0..127)

  const _Float16* Q  = Qh + (size_t)bh * NSEQ * DK + (size_t)qt * 16 * DK;
  const _Float16* Kp = Kh + (size_t)bh * NSEQ * DK;
  const _Float16* Vp = Vt + (size_t)bh * DK * NSEQ;
  _Float16* pb = psh + wave * (16 * 32);

  // K staging: thread -> K row tid/4 (of 32), 16-half slice (tid%4)*16
  int krow = tid >> 2, kcol = (tid & 3) * 16;
  const _Float16* ksrc = Kp + (size_t)krow * DK + kcol;
  _Float16*       kdst = ksh + krow * KSTRIDE + kcol;
  // V staging: thread -> dv row tid/2 (of 64), 16-half slice (tid%2)*16
  int vrow = tid >> 1, vcol = (tid & 1) * 16;
  const _Float16* vsrc = Vp + (size_t)vrow * NSEQ + vcol;
  _Float16*       vdst = vsh + vrow * BSTRIDE + vcol;

  // Q A-fragments for K-steps kb = 0, 32 of the dk=64 reduction
  AFrag qa[2];
  {
    const _Float16* qrow = Q + (size_t)l16 * DK;
#pragma unroll
    for (int j = 0; j < 2; ++j) {
      qa[j].h[0] = *(const v8h*)(qrow + j * 32 + hi * 16);
      qa[j].h[1] = *(const v8h*)(qrow + j * 32 + hi * 16 + 8);
    }
  }

  v8f o[4] = {};                                  // O: 16 x 64 (4 N-tiles)
  float rmax[8], rsum[8];
#pragma unroll
  for (int r = 0; r < 8; ++r) { rmax[r] = -1e30f; rsum[r] = 0.f; }

  for (int kv = 0; kv < NSEQ; kv += 32) {
    // ---- cooperative staging of this chunk's K and V (loads issued before
    //      the barrier so they overlap the previous chunk's compute) ----
    AFrag kstg, vstg;
    kstg.h[0] = *(const v8h*)(ksrc + (size_t)kv * DK);
    kstg.h[1] = *(const v8h*)(ksrc + (size_t)kv * DK + 8);
    vstg.h[0] = *(const v8h*)(vsrc + kv);
    vstg.h[1] = *(const v8h*)(vsrc + kv + 8);
    __syncthreads();                        // previous chunk fully consumed
    *(v8h*)(kdst)     = kstg.h[0];
    *(v8h*)(kdst + 8) = kstg.h[1];
    *(v8h*)(vdst)     = vstg.h[0];
    *(v8h*)(vdst + 8) = vstg.h[1];
    __syncthreads();                        // chunk visible to all waves

    // ---- S chunk = Q(16x64) * K^T  -> two 16x16 tiles (kv cols 0-15 / 16-31)
    v8f s0 = {}, s1 = {};
#pragma unroll
    for (int j = 0; j < 2; ++j) {
      const _Float16* kls0 = ksh + l16 * KSTRIDE + j * 32 + hi * 16;
      const _Float16* kls1 = kls0 + 16 * KSTRIDE;
      AFrag b0, b1;
      b0.h[0] = *(const v8h*)(kls0);  b0.h[1] = *(const v8h*)(kls0 + 8);
      b1.h[0] = *(const v8h*)(kls1);  b1.h[1] = *(const v8h*)(kls1 + 8);
      s0 = wmma16x16x32(qa[j].v, b0.v, s0);
      s1 = wmma16x16x32(qa[j].v, b1.v, s1);
    }

    // ---- online softmax per VGPR slot (row M = r + 8*hi) ----
#pragma unroll
    for (int r = 0; r < 8; ++r) {
      float m = fmaxf(s0[r], s1[r]);
#pragma unroll
      for (int off = 1; off < 16; off <<= 1)
        m = fmaxf(m, __shfl_xor(m, off, 32));    // reduce across N (16 lanes)
      float nm   = fmaxf(rmax[r], m);
      float corr = __expf(rmax[r] - nm);
      rmax[r] = nm;
      float p0 = __expf(s0[r] - nm);
      float p1 = __expf(s1[r] - nm);
      float ps = p0 + p1;
#pragma unroll
      for (int off = 1; off < 16; off <<= 1)
        ps += __shfl_xor(ps, off, 32);
      rsum[r] = rsum[r] * corr + ps;
#pragma unroll
      for (int t = 0; t < 4; ++t) o[t][r] *= corr;
      // stage P (row-major 16x32) into per-wave LDS for A-fragment reload
      pb[(r + hi * 8) * 32 + l16]      = (_Float16)p0;
      pb[(r + hi * 8) * 32 + l16 + 16] = (_Float16)p1;
    }

    // ---- O += P(16x32) * V(32x64): P A-fragment + V B-fragments from LDS ---
    AFrag pa;
    pa.h[0] = *(const v8h*)(pb + l16 * 32 + hi * 16);
    pa.h[1] = *(const v8h*)(pb + l16 * 32 + hi * 16 + 8);
#pragma unroll
    for (int t = 0; t < 4; ++t) {
      const _Float16* vls = vsh + (t * 16 + l16) * BSTRIDE + hi * 16;
      AFrag bv;
      bv.h[0] = *(const v8h*)(vls);
      bv.h[1] = *(const v8h*)(vls + 8);
      o[t] = wmma16x16x32(pa.v, bv.v, o[t]);
    }
  }

  // ---- normalize and write ctx [b][seq][h*64 + dv] as f16 ----
  int b = bh >> 4, h = bh & 15;
#pragma unroll
  for (int r = 0; r < 8; ++r) {
    float inv = 1.0f / rsum[r];
    int seq = qt * 16 + r + hi * 8;
#pragma unroll
    for (int t = 0; t < 4; ++t) {
      int dv = t * 16 + l16;
      size_t idx = ((size_t)b * NSEQ + seq) * D_MODEL + h * DK + dv;
      ctx[idx] = (_Float16)(o[t][r] * inv);
    }
  }
}

// ---------------------------------------------------------------------------
// 5) fc GEMM: preLN = relu(ctx @ w_fc) + residual(q)   (f32 out)
//    Same LDS-staged block tile as the projection GEMM.
// ---------------------------------------------------------------------------
__global__ __launch_bounds__(256)
void k_fc_gemm(const _Float16* __restrict__ A,     // ctx f16 [4096][1024]
               const _Float16* __restrict__ Bt,    // wfc_t  [1024][1024]
               const float* __restrict__ residual, // q f32
               float* __restrict__ preln) {
  __shared__ __align__(16) _Float16 bsh[64 * BSTRIDE];
  int tid  = threadIdx.x;
  int wave = tid >> 5;
  int lane = tid & 31;
  int hi   = lane >> 4;
  int l16  = lane & 15;
  int m0   = blockIdx.x * 128 + wave * 16;
  int n0   = blockIdx.y * 64;

  const _Float16* arow = A + (size_t)(m0 + l16) * D_MODEL;

  int srow = tid >> 2;
  int scol = (tid & 3) * 8;
  const _Float16* bsrc = Bt + (size_t)(n0 + srow) * D_MODEL + scol;
  _Float16*       bdst = bsh + srow * BSTRIDE + scol;

  v8f acc[4] = {};
  for (int kb = 0; kb < D_MODEL; kb += 32) {
    v8h stg = *(const v8h*)(bsrc + kb);
    __syncthreads();
    *(v8h*)bdst = stg;
    __syncthreads();

    AFrag a;
    a.h[0] = *(const v8h*)(arow + kb + hi * 16);
    a.h[1] = *(const v8h*)(arow + kb + hi * 16 + 8);
#pragma unroll
    for (int t = 0; t < 4; ++t) {
      const _Float16* bls = bsh + (t * 16 + l16) * BSTRIDE + hi * 16;
      AFrag bf;
      bf.h[0] = *(const v8h*)(bls);
      bf.h[1] = *(const v8h*)(bls + 8);
      acc[t] = wmma16x16x32(a.v, bf.v, acc[t]);
    }
  }

#pragma unroll
  for (int t = 0; t < 4; ++t) {
#pragma unroll
    for (int r = 0; r < 8; ++r) {
      int row = m0 + r + hi * 8;
      int col = n0 + t * 16 + l16;
      size_t idx = (size_t)row * D_MODEL + col;
      preln[idx] = fmaxf(acc[t][r], 0.0f) + residual[idx];
    }
  }
}

// ---------------------------------------------------------------------------
// 6) Row-wise LayerNorm over 1024, one 256-thread block per row
// ---------------------------------------------------------------------------
__global__ void k_layernorm(const float* __restrict__ x,
                            const float* __restrict__ gamma,
                            const float* __restrict__ beta,
                            float* __restrict__ out) {
  __shared__ float ws1[8], ws2[8];
  int row = blockIdx.x;
  const float* xr = x + (size_t)row * D_MODEL;
  float vals[4], s = 0.f, s2 = 0.f;
#pragma unroll
  for (int i = 0; i < 4; ++i) {
    float vv = xr[threadIdx.x + i * 256];
    vals[i] = vv; s += vv; s2 += vv * vv;
  }
#pragma unroll
  for (int off = 16; off >= 1; off >>= 1) {
    s  += __shfl_xor(s,  off, 32);
    s2 += __shfl_xor(s2, off, 32);
  }
  int wv = threadIdx.x >> 5, ln = threadIdx.x & 31;
  if (ln == 0) { ws1[wv] = s; ws2[wv] = s2; }
  __syncthreads();
  if (wv == 0) {
    s  = (ln < 8) ? ws1[ln] : 0.f;
    s2 = (ln < 8) ? ws2[ln] : 0.f;
#pragma unroll
    for (int off = 4; off >= 1; off >>= 1) {
      s  += __shfl_xor(s,  off, 32);
      s2 += __shfl_xor(s2, off, 32);
    }
    if (ln == 0) { ws1[0] = s; ws2[0] = s2; }
  }
  __syncthreads();
  float mean = ws1[0] * (1.0f / D_MODEL);
  float var  = ws2[0] * (1.0f / D_MODEL) - mean * mean;
  float rstd = rsqrtf(var + 1e-6f);
#pragma unroll
  for (int i = 0; i < 4; ++i) {
    int c = threadIdx.x + i * 256;
    out[(size_t)row * D_MODEL + c] = (vals[i] - mean) * rstd * gamma[c] + beta[c];
  }
}

// ---------------------------------------------------------------------------
// launch
// ---------------------------------------------------------------------------
extern "C" void kernel_launch(void* const* d_in, const int* in_sizes, int n_in,
                              void* d_out, int out_size, void* d_ws, size_t ws_size,
                              hipStream_t stream) {
  const float* q     = (const float*)d_in[0];
  const float* k     = (const float*)d_in[1];
  const float* v     = (const float*)d_in[2];
  const float* w_qs  = (const float*)d_in[3];
  const float* w_ks  = (const float*)d_in[4];
  const float* w_vs  = (const float*)d_in[5];
  const float* w_fc  = (const float*)d_in[6];
  const float* gamma = (const float*)d_in[7];
  const float* beta  = (const float*)d_in[8];
  float* out = (float*)d_out;

  char* ws = (char*)d_ws;
  size_t off = 0;
  auto alloc = [&](size_t bytes) -> void* {
    void* p = ws + off;
    off += (bytes + 255) & ~(size_t)255;
    return p;
  };
  const size_t XB = (size_t)MTOT * D_MODEL;   // 4096*1024
  const size_t WB = (size_t)D_MODEL * D_MODEL;
  _Float16* qf16  = (_Float16*)alloc(XB * 2);
  _Float16* kf16  = (_Float16*)alloc(XB * 2);
  _Float16* vf16  = (_Float16*)alloc(XB * 2);
  _Float16* wqs_t = (_Float16*)alloc(WB * 2);
  _Float16* wks_t = (_Float16*)alloc(WB * 2);
  _Float16* wvs_t = (_Float16*)alloc(WB * 2);
  _Float16* wfc_t = (_Float16*)alloc(WB * 2);
  _Float16* Qh    = (_Float16*)alloc(XB * 2); // [b,h,n,64] prescaled by 1/8
  _Float16* Kh    = (_Float16*)alloc(XB * 2); // [b,h,n,64]
  _Float16* Vt    = (_Float16*)alloc(XB * 2); // [b,h,64,n]
  _Float16* ctx   = (_Float16*)alloc(XB * 2); // [b,n,1024]
  float*    preln = (float*)alloc(XB * 4);

  // 1) converts
  int nElem = (int)XB;
  k_cvt_f16<<<nElem / 256, 256, 0, stream>>>(q, qf16, nElem);
  k_cvt_f16<<<nElem / 256, 256, 0, stream>>>(k, kf16, nElem);
  k_cvt_f16<<<nElem / 256, 256, 0, stream>>>(v, vf16, nElem);

  // 2) weight transposes (K=N=1024 -> 32x32 grid of 32x32 tiles)
  dim3 tg(D_MODEL / 32, D_MODEL / 32);
  k_transpose_cvt<<<tg, 256, 0, stream>>>(w_qs, wqs_t, D_MODEL, D_MODEL);
  k_transpose_cvt<<<tg, 256, 0, stream>>>(w_ks, wks_t, D_MODEL, D_MODEL);
  k_transpose_cvt<<<tg, 256, 0, stream>>>(w_vs, wvs_t, D_MODEL, D_MODEL);
  k_transpose_cvt<<<tg, 256, 0, stream>>>(w_fc, wfc_t, D_MODEL, D_MODEL);

  // 3) projections (8 waves/block, block tile 128x64, LDS-staged B)
  dim3 pg(MTOT / 128, D_MODEL / 64);
  k_proj_gemm<<<pg, 256, 0, stream>>>(qf16, wqs_t, Qh, 0.125f, 0); // /sqrt(64)
  k_proj_gemm<<<pg, 256, 0, stream>>>(kf16, wks_t, Kh, 1.0f, 0);
  k_proj_gemm<<<pg, 256, 0, stream>>>(vf16, wvs_t, Vt, 1.0f, 1);

  // 4) flash attention: 4 q-tiles per block share LDS-staged K/V chunks
  dim3 ag(NSEQ / 16 / 4, BATCH * N_HEAD);
  k_attention<<<ag, 128, 0, stream>>>(Qh, Kh, Vt, ctx);

  // 5) output projection + relu + residual
  k_fc_gemm<<<pg, 256, 0, stream>>>(ctx, wfc_t, q, preln);

  // 6) layernorm
  k_layernorm<<<MTOT, 256, 0, stream>>>(preln, gamma, beta, out);
}
